// CrossAttn_50775103373813
// MI455X (gfx1250) — compile-verified, matrix operand order checked by reading, not run on hardware
//
#include <hip/hip_runtime.h>

#define NH    8
#define DIMK  512
#define NPTS  1024
#define MNP   4096
#define BS    4
#define TOPK  32
#define DH    64

typedef __attribute__((ext_vector_type(16))) __bf16 v16bf;
typedef __attribute__((ext_vector_type(8)))  __bf16 v8bf;
typedef __attribute__((ext_vector_type(8)))  float  v8f;

union Frag { v16bf v; v8bf h[2]; };

__device__ __forceinline__ __bf16 f2bf(float f) {
  union { float f; unsigned u; } in; in.f = f;
  unsigned u = in.u;
  u += 0x7FFFu + ((u >> 16) & 1u);          // round-to-nearest-even
  union { unsigned short s; __bf16 b; } out;
  out.s = (unsigned short)(u >> 16);
  return out.b;
}

// Async 16-byte global->LDS copy (VFLAT async, tracked by ASYNCcnt).
// %0 = VGPR holding LDS byte offset, %1 = VGPR pair holding global address.
__device__ __forceinline__ void async_b128(unsigned lds_off, const void* gaddr) {
  asm volatile("global_load_async_to_lds_b128 %0, %1, off"
               :: "v"(lds_off), "v"((unsigned long long)(uintptr_t)gaddr)
               : "memory");
}
__device__ __forceinline__ unsigned lds_off32(const void* p) {
  // LDS aperture: addr[31:0] is the LDS byte offset (flat->LDS truncation).
  return (unsigned)(uintptr_t)p;
}

// ---------------------------------------------------------------------------
// Elementwise f32 -> bf16 (8 elems/thread, b128 in / b128 out).
// ---------------------------------------------------------------------------
__global__ __launch_bounds__(256)
void cvt_f32_bf16(const float* __restrict__ src, __bf16* __restrict__ dst, int n8) {
  int i = blockIdx.x * 256 + threadIdx.x;
  if (i >= n8) return;
  float4 a = ((const float4*)src)[2 * i];
  float4 b = ((const float4*)src)[2 * i + 1];
  v8bf o;
  o[0] = f2bf(a.x); o[1] = f2bf(a.y); o[2] = f2bf(a.z); o[3] = f2bf(a.w);
  o[4] = f2bf(b.x); o[5] = f2bf(b.y); o[6] = f2bf(b.z); o[7] = f2bf(b.w);
  ((v8bf*)dst)[i] = o;
}

// ---------------------------------------------------------------------------
// C[r, e] = sum_d A[r, d] * W[e, d] + bias[e]   (A, W already bf16)
// Block tile 128(M) x 64(N), K-step 32, 8 waves, each 32x32 (2x2 WMMA accs).
// Double-buffered LDS tiles filled by GLOBAL_LOAD_ASYNC_TO_LDS_B128.
// ---------------------------------------------------------------------------
#define TM 128
#define TN 64
#define TK 32
#define LDT (TK + 8)   // 40 halfs = 80B row stride (16B-aligned chunks)

__global__ __launch_bounds__(256)
void gemm_bias_bf16wmma(const __bf16* __restrict__ A,
                        const __bf16* __restrict__ W,
                        const float* __restrict__ bias,
                        float* __restrict__ C) {
  __shared__ __bf16 lA[2][TM][LDT];
  __shared__ __bf16 lB[2][TN][LDT];

  const int t     = threadIdx.x;
  const int lane  = t & 31;
  const int wid   = t >> 5;
  const int waveM = wid & 3;
  const int waveN = wid >> 2;
  const int rowBase = blockIdx.x * TM;
  const int colBase = blockIdx.y * TN;

  const int lrow = lane & 15;
  const int k0   = (lane < 16) ? 0 : 8;     // ISA 16-bit fragment layout

  // Per-thread staging assignment: 16B chunks (8 bf16), 4 chunks per 32-elem row.
  const int ra0 = t >> 2,          ca0 = (t & 3) << 3;         // A chunk 0
  const int ra1 = (t + 256) >> 2,  ca1 = ((t + 256) & 3) << 3; // A chunk 1
  const int rb  = t >> 2,          cb  = (t & 3) << 3;         // B chunk

  v8f acc[2][2] = {};

#define STAGE(buf, kb)                                                         \
  do {                                                                         \
    async_b128(lds_off32(&lA[buf][ra0][ca0],),                                 \
               A + (size_t)(rowBase + ra0) * DIMK + (kb) + ca0);               \
  } while (0)

#undef STAGE
  // (macro scrapped: write staging inline for clarity)

  // Prologue: stage buffer 0 for kb = 0.
  async_b128(lds_off32(&lA[0][ra0][ca0]), A + (size_t)(rowBase + ra0) * DIMK + ca0);
  async_b128(lds_off32(&lA[0][ra1][ca1]), A + (size_t)(rowBase + ra1) * DIMK + ca1);
  async_b128(lds_off32(&lB[0][rb ][cb ]), W + (size_t)(colBase + rb ) * DIMK + cb );

  const int NIT = DIMK / TK;     // 16
  for (int i = 0; i < NIT; ++i) {
    const int cur = i & 1;
    if (i + 1 < NIT) {
      const int nxt = cur ^ 1;
      const int kb  = (i + 1) * TK;
      async_b128(lds_off32(&lA[nxt][ra0][ca0]), A + (size_t)(rowBase + ra0) * DIMK + kb + ca0);
      async_b128(lds_off32(&lA[nxt][ra1][ca1]), A + (size_t)(rowBase + ra1) * DIMK + kb + ca1);
      async_b128(lds_off32(&lB[nxt][rb ][cb ]), W + (size_t)(colBase + rb ) * DIMK + kb + cb );
      // 6 outstanding; async loads retire in order -> <=3 means buf[cur] landed.
      asm volatile("s_wait_asynccnt 0x3" ::: "memory");
    } else {
      asm volatile("s_wait_asynccnt 0x0" ::: "memory");
    }
    __syncthreads();

    const __bf16 (*pA)[LDT] = lA[cur];
    const __bf16 (*pB)[LDT] = lB[cur];
    Frag a[2], bfr[2];
#pragma unroll
    for (int mi = 0; mi < 2; ++mi) {
      int ar = waveM * 32 + mi * 16 + lrow;
      a[mi].h[0] = *(const v8bf*)&pA[ar][k0];
      a[mi].h[1] = *(const v8bf*)&pA[ar][k0 + 16];
    }
#pragma unroll
    for (int ni = 0; ni < 2; ++ni) {
      int br = waveN * 32 + ni * 16 + lrow;
      bfr[ni].h[0] = *(const v8bf*)&pB[br][k0];
      bfr[ni].h[1] = *(const v8bf*)&pB[br][k0 + 16];
    }
#pragma unroll
    for (int mi = 0; mi < 2; ++mi)
#pragma unroll
      for (int ni = 0; ni < 2; ++ni)
        acc[mi][ni] = __builtin_amdgcn_wmma_f32_16x16x32_bf16(
            false, a[mi].v, false, bfr[ni].v,
            (short)0, acc[mi][ni], false, false);
    __syncthreads();   // reads of buf[cur] done before it is re-staged
  }

  // Epilogue: C/D layout -> lane l, vgpr j: M = j + 8*(l>=16), N = l%16.
  const int rhalf = (lane < 16) ? 0 : 8;
#pragma unroll
  for (int mi = 0; mi < 2; ++mi)
#pragma unroll
    for (int ni = 0; ni < 2; ++ni) {
      int col = colBase + waveN * 32 + ni * 16 + lrow;
      float bv = bias[col];
#pragma unroll
      for (int j = 0; j < 8; ++j) {
        int row = rowBase + waveM * 32 + mi * 16 + j + rhalf;
        C[(size_t)row * DIMK + col] = acc[mi][ni][j] + bv;
      }
    }
}

// ---------------------------------------------------------------------------
// Per-(b,n) top-k attention: dedupe indices, masked softmax over <=32 keys,
// weighted sum of vh rows. Wave w == head w (wave32 shuffle reductions).
// ---------------------------------------------------------------------------
__global__ __launch_bounds__(256)
void attn_topk(const float* __restrict__ qh, const float* __restrict__ kh,
               const float* __restrict__ vh, const int* __restrict__ rns,
               float* __restrict__ out) {
  const int n = blockIdx.x;
  const int b = blockIdx.y;

  __shared__ float sq[DIMK];
  __shared__ int   sidx[TOPK];
  __shared__ float svalid[TOPK];
  __shared__ float sw[NH][TOPK];

  const int t    = threadIdx.x;
  const int lane = t & 31;
  const int h    = t >> 5;
  const size_t qrow = (size_t)(n * BS + b) * DIMK;

  sq[t]       = qh[qrow + t];
  sq[t + 256] = qh[qrow + t + 256];
  if (t < TOPK) sidx[t] = rns[((size_t)b * NPTS + n) * TOPK + t];
  __syncthreads();

  if (t < TOPK) {             // duplicates count once (mask .set semantics)
    float v = 1.0f;
    for (int j = 0; j < t; ++j) if (sidx[j] == sidx[t]) v = 0.0f;
    svalid[t] = v;
  }
  __syncthreads();

  const float* kr = kh + ((size_t)sidx[lane] * BS + b) * DIMK + h * DH;
  const float* qr = sq + h * DH;
  float acc = 0.0f;
#pragma unroll 8
  for (int d = 0; d < DH; ++d) acc += qr[d] * kr[d];
  acc *= 0.125f;                                  // 1/sqrt(64)
  const bool ok = svalid[lane] > 0.0f;

  float m = ok ? acc : -3.0e38f;
#pragma unroll
  for (int off = 16; off > 0; off >>= 1) m = fmaxf(m, __shfl_xor(m, off));
  float e = ok ? __expf(acc - m) : 0.0f;
  float s = e;
#pragma unroll
  for (int off = 16; off > 0; off >>= 1) s += __shfl_xor(s, off);
  sw[h][lane] = e / s;
  __syncthreads();

#pragma unroll
  for (int rep = 0; rep < 2; ++rep) {
    int eo = t + rep * 256;
    int h2 = eo / DH;
    float o = 0.0f;
#pragma unroll 8
    for (int i = 0; i < TOPK; ++i)
      o += sw[h2][i] * vh[((size_t)sidx[i] * BS + b) * DIMK + eo];
    out[(size_t)(n * BS + b) * DIMK + eo] = o;
  }
}

// ---------------------------------------------------------------------------
extern "C" void kernel_launch(void* const* d_in, const int* in_sizes, int n_in,
                              void* d_out, int out_size, void* d_ws, size_t ws_size,
                              hipStream_t stream) {
  const float* q   = (const float*)d_in[0];
  const float* k   = (const float*)d_in[1];
  const float* v   = (const float*)d_in[2];
  const int*   rns = (const int*)  d_in[3];
  const float* W1  = (const float*)d_in[4];
  const float* b1  = (const float*)d_in[5];
  const float* W2  = (const float*)d_in[6];
  const float* b2  = (const float*)d_in[7];
  const float* W3  = (const float*)d_in[8];
  const float* b3  = (const float*)d_in[9];
  float* out = (float*)d_out;

  const size_t nQ = (size_t)NPTS * BS * DIMK;   // 2M
  const size_t nK = (size_t)MNP  * BS * DIMK;   // 8M
  const size_t nW = (size_t)DIMK * DIMK;        // 256K

  // Workspace: f32 projections first, then bf16 staging copies.
  float*  qh  = (float*)d_ws;
  float*  kh  = qh + nQ;
  float*  vh  = kh + nK;
  __bf16* qb  = (__bf16*)(vh + nK);
  __bf16* kb  = qb + nQ;
  __bf16* vb  = kb + nK;
  __bf16* w1b = vb + nK;
  __bf16* w2b = w1b + nW;
  __bf16* w3b = w2b + nW;

  dim3 blk(256);
  cvt_f32_bf16<<<dim3((unsigned)(nQ / 8 / 256)), blk, 0, stream>>>(q,  qb,  (int)(nQ / 8));
  cvt_f32_bf16<<<dim3((unsigned)(nK / 8 / 256)), blk, 0, stream>>>(k,  kb,  (int)(nK / 8));
  cvt_f32_bf16<<<dim3((unsigned)(nK / 8 / 256)), blk, 0, stream>>>(v,  vb,  (int)(nK / 8));
  cvt_f32_bf16<<<dim3((unsigned)(nW / 8 / 256)), blk, 0, stream>>>(W1, w1b, (int)(nW / 8));
  cvt_f32_bf16<<<dim3((unsigned)(nW / 8 / 256)), blk, 0, stream>>>(W2, w2b, (int)(nW / 8));
  cvt_f32_bf16<<<dim3((unsigned)(nW / 8 / 256)), blk, 0, stream>>>(W3, w3b, (int)(nW / 8));

  gemm_bias_bf16wmma<<<dim3((NPTS * BS) / TM, DIMK / TN), blk, 0, stream>>>(qb, w1b, b1, qh);
  gemm_bias_bf16wmma<<<dim3((MNP  * BS) / TM, DIMK / TN), blk, 0, stream>>>(kb, w2b, b2, kh);
  gemm_bias_bf16wmma<<<dim3((MNP  * BS) / TM, DIMK / TN), blk, 0, stream>>>(vb, w3b, b3, vh);
  attn_topk<<<dim3(NPTS, BS), blk, 0, stream>>>(qh, kh, vh, rns, out);
}
